// GlimpseNetwork_35897336660352
// MI455X (gfx1250) — compile-verified, matrix operand order checked by reading, not run on hardware
//
#include <hip/hip_runtime.h>

#define BATCH   4096
#define HIMG    128
#define G14     14
#define PHI     784
#define PHI_PAD 800     // 25 * 32, K padded for WMMA
#define H1      128
#define H2      256
#define BN_EPS  1e-5f

typedef __attribute__((ext_vector_type(16))) _Float16 v16h;
typedef __attribute__((ext_vector_type(8)))  _Float16 v8h;
typedef __attribute__((ext_vector_type(8)))  float    v8f;

// ---------------------------------------------------------------------------
// Foveation: one block per image. Each thread (tid<196) owns one 14x14 output
// pixel and computes it for all 4 scales (pool k = 1,2,4,8). Zero outside the
// image == the reference's zero-padding; divide by k*k always (matches
// avg_pool2d over the zero-padded patch). Writes f16 phi, K-padded to 800.
// ---------------------------------------------------------------------------
__global__ void foveate_kernel(const float* __restrict__ x,
                               const float* __restrict__ loc,
                               _Float16* __restrict__ phi16) {
  const int b   = blockIdx.x;
  const int tid = threadIdx.x;
  const float lx = loc[b * 2 + 0];
  const float ly = loc[b * 2 + 1];
  const int cx = (int)(0.5f * ((lx + 1.0f) * (float)HIMG));
  const int cy = (int)(0.5f * ((ly + 1.0f) * (float)HIMG));
  const float* img = x + (size_t)b * HIMG * HIMG;

  if (tid < 196) {
    const int i = tid / 14;
    const int j = tid % 14;
    int size = G14;
#pragma unroll
    for (int s = 0; s < 4; ++s) {
      const int k  = size / G14;
      const int x0 = cx - size / 2;
      const int y0 = cy - size / 2;
      float sum = 0.0f;
      for (int dy = 0; dy < k; ++dy) {
        const int yy = y0 + i * k + dy;
        if (yy < 0 || yy >= HIMG) continue;
        const float* row = img + yy * HIMG;
        for (int dx = 0; dx < k; ++dx) {
          const int xx = x0 + j * k + dx;
          if (xx < 0 || xx >= HIMG) continue;
          sum += row[xx];
        }
      }
      const float avg = sum / (float)(k * k);
      phi16[(size_t)b * PHI_PAD + s * 196 + i * 14 + j] = (_Float16)avg;
      size *= 2;
    }
  }
  // zero the K-padding columns 784..799
  if (tid < (PHI_PAD - PHI)) {
    phi16[(size_t)b * PHI_PAD + PHI + tid] = (_Float16)0.0f;
  }
}

// ---------------------------------------------------------------------------
// Weight pre-pack: rearrange B[K,N] fp32 into per-lane WMMA fragment order,
// f16.  Packed layout: tile (kt,nt) -> 512 halfs; lane L owns halfs
// [L*16 .. L*16+15] = its 16x16x32-f16 B-fragment in VGPR order.  K-padding
// rows (>= Ksrc) become zeros.  One thread per (tile, lane).
// ---------------------------------------------------------------------------
__global__ void pack_b_kernel(const float* __restrict__ B,
                              _Float16* __restrict__ Bp,
                              int K, int N, int Ksrc) {
  const int t = blockIdx.x * blockDim.x + threadIdx.x;
  const int tilesN = N >> 4;
  const int tilesK = K >> 5;
  if (t >= tilesK * tilesN * 32) return;
  const int lane = t & 31;
  const int tile = t >> 5;
  const int kt = tile / tilesN;
  const int nt = tile % tilesN;
  const int r16  = lane & 15;   // N column within tile
  const int hsel = lane >> 4;
  const int n = nt * 16 + r16;
  _Float16* o = Bp + (size_t)tile * 512 + lane * 16;
#pragma unroll
  for (int i = 0; i < 8; ++i) {
    const int kk = kt * 32 + hsel * 8 + ((i < 4) ? (2 * i) : (16 + 2 * (i - 4)));
    o[2 * i]     = (kk     < Ksrc) ? (_Float16)B[(size_t)kk       * N + n] : (_Float16)0.0f;
    o[2 * i + 1] = (kk + 1 < Ksrc) ? (_Float16)B[(size_t)(kk + 1) * N + n] : (_Float16)0.0f;
  }
}

// ---------------------------------------------------------------------------
// WMMA GEMM + bias:  C[M,N] f32 = A[M,K] f16 @ B(packed) f16 + bias[N]
// One wave computes a 16(M) x 64(N) strip: 4 accumulators share one A
// fragment per k-step -> 4 v_wmma per iteration, A loaded once.
// A fragment: two contiguous v8h (b128) chunks per lane (ISA 7.12.2 layout).
// B fragment: one contiguous v16h per lane from the packed weights.
// Wave-uniform early exit keeps EXEC all-1s for every WMMA.
// ---------------------------------------------------------------------------
__global__ void wmma_gemm_bias_kernel(const _Float16* __restrict__ A,
                                      const _Float16* __restrict__ Bp,
                                      const float* __restrict__ bias,
                                      float* __restrict__ Cm,
                                      int M, int N, int K) {
  const int wavesPerBlock = blockDim.x >> 5;
  const int wave    = blockIdx.x * wavesPerBlock + ((int)threadIdx.x >> 5);
  const int tilesN  = N >> 4;
  const int tilesN4 = N >> 6;   // groups of 4 N-tiles
  const int tilesM  = M >> 4;
  if (wave >= tilesM * tilesN4) return;         // wave-uniform exit
  const int tm  = (wave / tilesN4) << 4;
  const int ng  = (wave % tilesN4) * 4;         // first N-tile of this strip
  const int lane = threadIdx.x & 31;
  const int r16  = lane & 15;
  const int hsel = lane >> 4;

  v8f acc0 = {}, acc1 = {}, acc2 = {}, acc3 = {};
  const _Float16* Arow = A + (size_t)(tm + r16) * K + hsel * 8;
  const _Float16* Bl   = Bp + (size_t)lane * 16;

  for (int k0 = 0; k0 < K; k0 += 32) {
    __builtin_prefetch(Arow + k0 + 32, 0, 0);   // global_prefetch_b8
    const v8h a_lo = *(const v8h*)(Arow + k0);
    const v8h a_hi = *(const v8h*)(Arow + k0 + 16);
    const v16h af = __builtin_shufflevector(a_lo, a_hi,
        0, 1, 2, 3, 4, 5, 6, 7, 8, 9, 10, 11, 12, 13, 14, 15);
    const size_t kbase = (size_t)(k0 >> 5) * tilesN * 512;
    const v16h bf0 = *(const v16h*)(Bl + kbase + (size_t)(ng + 0) * 512);
    const v16h bf1 = *(const v16h*)(Bl + kbase + (size_t)(ng + 1) * 512);
    const v16h bf2 = *(const v16h*)(Bl + kbase + (size_t)(ng + 2) * 512);
    const v16h bf3 = *(const v16h*)(Bl + kbase + (size_t)(ng + 3) * 512);
    acc0 = __builtin_amdgcn_wmma_f32_16x16x32_f16(false, af, false, bf0, (short)0, acc0, false, false);
    acc1 = __builtin_amdgcn_wmma_f32_16x16x32_f16(false, af, false, bf1, (short)0, acc1, false, false);
    acc2 = __builtin_amdgcn_wmma_f32_16x16x32_f16(false, af, false, bf2, (short)0, acc2, false, false);
    acc3 = __builtin_amdgcn_wmma_f32_16x16x32_f16(false, af, false, bf3, (short)0, acc3, false, false);
  }

#pragma unroll
  for (int t = 0; t < 4; ++t) {
    const v8f acc = (t == 0) ? acc0 : (t == 1) ? acc1 : (t == 2) ? acc2 : acc3;
    const int tn = (ng + t) << 4;
    const float bv = bias[tn + r16];
#pragma unroll
    for (int r = 0; r < 8; ++r) {
      const int row = tm + r + hsel * 8;
      Cm[(size_t)row * N + tn + r16] = acc[r] + bv;
    }
  }
}

// ---------------------------------------------------------------------------
// where path layer 1: h[i,j] = loc[i,0]*W[0,j] + loc[i,1]*W[1,j] + b[j]
// ---------------------------------------------------------------------------
__global__ void where_l1_kernel(const float* __restrict__ loc,
                                const float* __restrict__ W,
                                const float* __restrict__ bv,
                                float* __restrict__ h) {
  const int idx = blockIdx.x * blockDim.x + threadIdx.x;
  if (idx >= BATCH * H1) return;
  const int i = idx / H1;
  const int j = idx % H1;
  h[idx] = loc[i * 2 + 0] * W[j] + loc[i * 2 + 1] * W[H1 + j] + bv[j];
}

// ---------------------------------------------------------------------------
// BatchNorm statistics (training mode, biased variance). One block per
// feature; stats[2j] = mean, stats[2j+1] = rsqrt(var + eps).
// ---------------------------------------------------------------------------
__global__ void bn_stats_kernel(const float* __restrict__ h,
                                float* __restrict__ stats, int R, int F) {
  const int j = blockIdx.x;
  __shared__ float ssum[256];
  __shared__ float ssq[256];
  float sum = 0.0f, sq = 0.0f;
  for (int r = threadIdx.x; r < R; r += blockDim.x) {
    const float v = h[(size_t)r * F + j];
    sum += v;
    sq  += v * v;
  }
  ssum[threadIdx.x] = sum;
  ssq[threadIdx.x]  = sq;
  __syncthreads();
  for (int s = blockDim.x >> 1; s > 0; s >>= 1) {
    if ((int)threadIdx.x < s) {
      ssum[threadIdx.x] += ssum[threadIdx.x + s];
      ssq[threadIdx.x]  += ssq[threadIdx.x + s];
    }
    __syncthreads();
  }
  if (threadIdx.x == 0) {
    const float m   = ssum[0] / (float)R;
    const float var = ssq[0] / (float)R - m * m;
    stats[2 * j]     = m;
    stats[2 * j + 1] = rsqrtf(var + BN_EPS);
  }
}

// Fused BN + ReLU + f16 convert (hidden activations feeding GEMM2)
__global__ void bn_relu_f16_kernel(const float* __restrict__ h,
                                   const float* __restrict__ stats,
                                   const float* __restrict__ g,
                                   const float* __restrict__ be,
                                   _Float16* __restrict__ o, int R, int F) {
  const int idx = blockIdx.x * blockDim.x + threadIdx.x;
  if (idx >= R * F) return;
  const int j = idx % F;
  const float v = (h[idx] - stats[2 * j]) * stats[2 * j + 1] * g[j] + be[j];
  o[idx] = (_Float16)fmaxf(v, 0.0f);
}

// Final: out = relu( BN(h2_what) + BN(h2_where) )
__global__ void final_kernel(const float* __restrict__ hw, const float* __restrict__ sw,
                             const float* __restrict__ gw, const float* __restrict__ bew,
                             const float* __restrict__ hk, const float* __restrict__ sk,
                             const float* __restrict__ gk, const float* __restrict__ bek,
                             float* __restrict__ out, int R, int F) {
  const int idx = blockIdx.x * blockDim.x + threadIdx.x;
  if (idx >= R * F) return;
  const int j = idx % F;
  const float a = (hw[idx] - sw[2 * j]) * sw[2 * j + 1] * gw[j] + bew[j];
  const float b = (hk[idx] - sk[2 * j]) * sk[2 * j + 1] * gk[j] + bek[j];
  out[idx] = fmaxf(a + b, 0.0f);
}

// ---------------------------------------------------------------------------
extern "C" void kernel_launch(void* const* d_in, const int* in_sizes, int n_in,
                              void* d_out, int out_size, void* d_ws, size_t ws_size,
                              hipStream_t stream) {
  const float* x    = (const float*)d_in[0];
  const float* loc  = (const float*)d_in[1];
  const float* wW1  = (const float*)d_in[2];
  const float* wb1  = (const float*)d_in[3];
  const float* wg1  = (const float*)d_in[4];
  const float* wbe1 = (const float*)d_in[5];
  const float* wW2  = (const float*)d_in[6];
  const float* wb2  = (const float*)d_in[7];
  const float* wg2  = (const float*)d_in[8];
  const float* wbe2 = (const float*)d_in[9];
  const float* rW1  = (const float*)d_in[10];
  const float* rb1  = (const float*)d_in[11];
  const float* rg1  = (const float*)d_in[12];
  const float* rbe1 = (const float*)d_in[13];
  const float* rW2  = (const float*)d_in[14];
  const float* rb2  = (const float*)d_in[15];
  const float* rg2  = (const float*)d_in[16];
  const float* rbe2 = (const float*)d_in[17];
  float* out = (float*)d_out;

  // packed weight tile counts
  const int w1Tiles = (PHI_PAD / 32) * (H1 / 16);   // 25 * 8  = 200
  const int w2Tiles = (H1 / 32) * (H2 / 16);        // 4 * 16  = 64

  // workspace carve-up (256B aligned)
  char* ws = (char*)d_ws;
  size_t off = 0;
  auto carve = [&](size_t bytes) -> void* {
    void* p = ws + off;
    off = (off + bytes + 255) & ~(size_t)255;
    return p;
  };
  _Float16* phi16   = (_Float16*)carve((size_t)BATCH * PHI_PAD * sizeof(_Float16));
  _Float16* w1p     = (_Float16*)carve((size_t)w1Tiles * 512 * sizeof(_Float16));
  _Float16* w2p     = (_Float16*)carve((size_t)w2Tiles * 512 * sizeof(_Float16));
  _Float16* rw2p    = (_Float16*)carve((size_t)w2Tiles * 512 * sizeof(_Float16));
  float*    h1w     = (float*)carve((size_t)BATCH * H1 * sizeof(float));
  float*    h1k     = (float*)carve((size_t)BATCH * H1 * sizeof(float));
  _Float16* a16w    = (_Float16*)carve((size_t)BATCH * H1 * sizeof(_Float16));
  _Float16* a16k    = (_Float16*)carve((size_t)BATCH * H1 * sizeof(_Float16));
  float*    h2w     = (float*)carve((size_t)BATCH * H2 * sizeof(float));
  float*    h2k     = (float*)carve((size_t)BATCH * H2 * sizeof(float));
  float*    stats1w = (float*)carve(2 * H1 * sizeof(float));
  float*    stats1k = (float*)carve(2 * H1 * sizeof(float));
  float*    stats2w = (float*)carve(2 * H2 * sizeof(float));
  float*    stats2k = (float*)carve(2 * H2 * sizeof(float));
  (void)ws_size; (void)in_sizes; (void)n_in; (void)out_size;

  // 1) foveation -> phi (f16, K-padded)
  foveate_kernel<<<BATCH, 256, 0, stream>>>(x, loc, phi16);

  // 2) weight pre-pack into WMMA fragment order
  pack_b_kernel<<<(w1Tiles * 32 + 255) / 256, 256, 0, stream>>>(
      wW1, w1p, PHI_PAD, H1, PHI);
  pack_b_kernel<<<(w2Tiles * 32 + 255) / 256, 256, 0, stream>>>(
      wW2, w2p, H1, H2, H1);
  pack_b_kernel<<<(w2Tiles * 32 + 255) / 256, 256, 0, stream>>>(
      rW2, rw2p, H1, H2, H1);

  // 3) GEMM1 (what): h1w = phi @ W1 + b1   [4096 x 800] x [800 x 128]
  {
    const int waves = (BATCH / 16) * (H1 / 64);   // 256 * 2 = 512
    wmma_gemm_bias_kernel<<<(waves + 7) / 8, 256, 0, stream>>>(
        phi16, w1p, wb1, h1w, BATCH, H1, PHI_PAD);
  }

  // 4) where layer 1 (K=2, VALU)
  where_l1_kernel<<<(BATCH * H1 + 255) / 256, 256, 0, stream>>>(loc, rW1, rb1, h1k);

  // 5) BN + ReLU -> f16 activations
  bn_stats_kernel<<<H1, 256, 0, stream>>>(h1w, stats1w, BATCH, H1);
  bn_relu_f16_kernel<<<(BATCH * H1 + 255) / 256, 256, 0, stream>>>(
      h1w, stats1w, wg1, wbe1, a16w, BATCH, H1);
  bn_stats_kernel<<<H1, 256, 0, stream>>>(h1k, stats1k, BATCH, H1);
  bn_relu_f16_kernel<<<(BATCH * H1 + 255) / 256, 256, 0, stream>>>(
      h1k, stats1k, rg1, rbe1, a16k, BATCH, H1);

  // 6) GEMM2 both paths: [4096 x 128] x [128 x 256]
  {
    const int waves = (BATCH / 16) * (H2 / 64);   // 256 * 4 = 1024
    wmma_gemm_bias_kernel<<<(waves + 7) / 8, 256, 0, stream>>>(
        a16w, w2p, wb2, h2w, BATCH, H2, H1);
    wmma_gemm_bias_kernel<<<(waves + 7) / 8, 256, 0, stream>>>(
        a16k, rw2p, rb2, h2k, BATCH, H2, H1);
  }

  // 7) output BN stats + fused add/relu
  bn_stats_kernel<<<H2, 256, 0, stream>>>(h2w, stats2w, BATCH, H2);
  bn_stats_kernel<<<H2, 256, 0, stream>>>(h2k, stats2k, BATCH, H2);
  final_kernel<<<(BATCH * H2 + 255) / 256, 256, 0, stream>>>(
      h2w, stats2w, wg2, wbe2, h2k, stats2k, rg2, rbe2, out, BATCH, H2);
}
